// SparseVLMModel_15109694948048
// MI455X (gfx1250) — matrix operand order repaired and checked by reading, not compiled, not run on hardware
//
#include <hip/hip_runtime.h>

// ---------------------------------------------------------------------------
// SparseVLM forward for MI455X (gfx1250, wave32).
// Memory/latency-bound (~100MB traffic @ 23.3 TB/s): bf16 WMMA GEMMs with
// TDM (tensor_load_to_lds) double-buffered tile staging, b128 LDS fragments.
// ---------------------------------------------------------------------------

constexpr int NB = 64;    // batch
constexpr int NP = 196;   // patch tokens
constexpr int NT = 256;   // text tokens
constexpr int ND = 768;   // model dim

typedef __attribute__((ext_vector_type(16))) __bf16        v16bf;
typedef __attribute__((ext_vector_type(8)))  float         v8f;
typedef __attribute__((ext_vector_type(4)))  unsigned int  u32x4;
typedef __attribute__((ext_vector_type(8)))  int           i32x8;
typedef __attribute__((ext_vector_type(4)))  int           i32x4;

// f32 -> bf16 (round to nearest even)
__device__ inline unsigned short f2bf(float f) {
  unsigned int u = __float_as_uint(f);
  unsigned int lsb = (u >> 16) & 1u;
  u += 0x7FFFu + lsb;
  return (unsigned short)(u >> 16);
}

// Generic pointer low 32 bits == LDS byte address (ISA 10.2 aperture rules).
__device__ inline unsigned lds_addr_of(const void* p) {
  return (unsigned)(unsigned long long)p;
}

// ---------------------------------------------------------------------------
// TDM: 2-D tile load global->LDS. data_size=2 bytes. Rows beyond tensor_d1
// zero-fill (OOB reads return zero). Row r of the tile lands at
// lds + r*tile_d0*2 bytes (contiguous), i.e. K-major rows of tile_d0 elems.
// D# layout per CDNA5 ISA.
// This toolchain exposes the 6-arg builtin:
//   (uint32x4 g0, int32x8 g1, int32x4 g2, int32x4 g3, int32x8 pad, i32 cpol)
// ---------------------------------------------------------------------------
__device__ inline void tdm_load_2d(unsigned lds_byte_addr, const void* gaddr,
                                   unsigned tensor_d0, unsigned tensor_d1,
                                   unsigned tile_d0, unsigned tile_d1,
                                   unsigned row_stride_elems) {
  unsigned long long ga = (unsigned long long)gaddr;
  u32x4 g0;
  g0[0] = 1u;                                        // count=1, user descriptor
  g0[1] = lds_byte_addr;                             // lds_addr
  g0[2] = (unsigned)(ga & 0xFFFFFFFFu);              // global_addr[31:0]
  g0[3] = (unsigned)((ga >> 32) & 0x1FFFFFFu) | (2u << 30);  // addr[56:32], type=2
  i32x8 g1;
  g1[0] = (int)(1u << 16);                           // data_size=1 (2B), mask=0
  g1[1] = (int)((tensor_d0 & 0xFFFFu) << 16);        // tensor_dim0[15:0]
  g1[2] = (int)(((tensor_d0 >> 16) & 0xFFFFu) | ((tensor_d1 & 0xFFFFu) << 16));
  g1[3] = (int)(((tensor_d1 >> 16) & 0xFFFFu) | ((tile_d0 & 0xFFFFu) << 16));
  g1[4] = (int)(tile_d1 & 0xFFFFu);                  // tile_dim1, tile_dim2=0
  g1[5] = (int)row_stride_elems;                     // tensor_dim0_stride[31:0]
  g1[6] = 0;                                         // stride0 hi, stride1 lo
  g1[7] = 0;
  i32x4 z4 = {0, 0, 0, 0};
  i32x8 z8 = {0, 0, 0, 0, 0, 0, 0, 0};
  __builtin_amdgcn_tensor_load_to_lds(g0, g1, z4, z4, z8, 0);
}

// ---------------------------------------------------------------------------
// WMMA fragments from K-major LDS tiles (rows of 32 bf16 = 64B, 16B aligned).
// A tile: 16 rows (M) x 32 (K).  B tile: N rows x 32 (K).
// Each fragment = two 16B LDS loads (ds_load_b128).
// ---------------------------------------------------------------------------
__device__ inline v16bf load_frag_a(const unsigned short* tile, int lane) {
  int m = lane & 15, half = lane >> 4;
  const u32x4* p = (const u32x4*)(tile + m * 32);
  union { v16bf v; u32x4 q[2]; } f;
  f.q[0] = p[half];       // elems kb..kb+7   (kb = 8*half)
  f.q[1] = p[2 + half];   // elems 16+kb..16+kb+7
  return f.v;
}

__device__ inline v16bf load_frag_b(const unsigned short* tile, int ncol, int lane) {
  int n = ncol + (lane & 15), half = lane >> 4;
  const u32x4* p = (const u32x4*)(tile + n * 32);
  union { v16bf v; u32x4 q[2]; } f;
  f.q[0] = p[half];       // K 8*half .. +7
  f.q[1] = p[2 + half];   // K 16+8*half .. +7
  return f.v;
}

// shared-memory argmax reduce over 256 slots (max value, ties -> lowest index)
__device__ inline int argmax_reduce(float* rbuf, int* ibuf, int tid) {
  for (int s = 128; s > 0; s >>= 1) {
    if (tid < s) {
      if (rbuf[tid + s] > rbuf[tid] ||
          (rbuf[tid + s] == rbuf[tid] && ibuf[tid + s] < ibuf[tid])) {
        rbuf[tid] = rbuf[tid + s];
        ibuf[tid] = ibuf[tid + s];
      }
    }
    __syncthreads();
  }
  return ibuf[0];
}

// ---------------------------------------------------------------------------
// Utility kernels
// ---------------------------------------------------------------------------
__global__ void zero_kernel(float* p, size_t n) {
  size_t i = (size_t)blockIdx.x * 256u + threadIdx.x;
  if (i < n) p[i] = 0.0f;
}

// f32 -> packed bf16 pairs
__global__ void cvt_bf16_kernel(const float* __restrict__ in,
                                unsigned int* __restrict__ out, size_t npair) {
  size_t i = (size_t)blockIdx.x * 256u + threadIdx.x;
  if (i < npair) {
    float x = in[2 * i], y = in[2 * i + 1];
    out[i] = (unsigned)f2bf(x) | ((unsigned)f2bf(y) << 16);
  }
}

// ---------------------------------------------------------------------------
// Kernel 1: attention GEMM (bf16 WMMA, TDM-staged, double buffered) + fused
// softmax over T; accumulates relevance column sums into rel[b][t].
// grid = (B, ceil(NP/16)); block = 256 (8 waves; wave w owns cols [32w,32w+32))
// ---------------------------------------------------------------------------
__global__ __launch_bounds__(256) void attn_kernel(const unsigned short* __restrict__ patchb,
                                                   const unsigned short* __restrict__ textb,
                                                   float* __restrict__ smbuf,
                                                   float* __restrict__ rel) {
  int b = blockIdx.x;
  int m0 = blockIdx.y * 16;
  int tid = threadIdx.x;
  int wave = tid >> 5, lane = tid & 31;

  __shared__ alignas(16) unsigned short aS[2][16 * 32];
  __shared__ alignas(16) unsigned short bS[2][256 * 32];
  __shared__ float sc[16][256];
  __shared__ float red[256];

  const unsigned short* pA = patchb + (size_t)(b * NP + m0) * ND;
  const unsigned short* pB = textb + (size_t)b * NT * ND;
  const int NSTG = ND / 32;  // 24

  v8f acc0 = {}, acc1 = {};

  if (tid == 0) {
    tdm_load_2d(lds_addr_of(aS[0]), pA, ND, NP - m0, 32, 16, ND);
    tdm_load_2d(lds_addr_of(bS[0]), pB, ND, NT, 32, 256, ND);
  }
  for (int s = 0; s < NSTG; ++s) {
    int cur = s & 1;
    if (tid == 0) {
      if (s + 1 < NSTG) {
        int k0 = 32 * (s + 1);
        tdm_load_2d(lds_addr_of(aS[1 - cur]), pA + k0, ND - k0, NP - m0, 32, 16, ND);
        tdm_load_2d(lds_addr_of(bS[1 - cur]), pB + k0, ND - k0, NT, 32, 256, ND);
        __builtin_amdgcn_s_wait_tensorcnt((short)2);
      } else {
        __builtin_amdgcn_s_wait_tensorcnt((short)0);
      }
    }
    __syncthreads();
    v16bf a  = load_frag_a(aS[cur], lane);
    v16bf b0 = load_frag_b(bS[cur], wave * 32, lane);
    v16bf b1 = load_frag_b(bS[cur], wave * 32 + 16, lane);
    acc0 = __builtin_amdgcn_wmma_f32_16x16x32_bf16(false, a, false, b0, (short)0, acc0, false, false);
    acc1 = __builtin_amdgcn_wmma_f32_16x16x32_bf16(false, a, false, b1, (short)0, acc1, false, false);
    __syncthreads();
  }

  // dump accumulators: VGPR r -> row r (lanes 0-15) / r+8 (lanes 16-31)
  int half = lane >> 4, col = lane & 15;
#pragma unroll
  for (int r = 0; r < 8; ++r) {
    sc[r + 8 * half][wave * 32 + col]      = acc0[r];
    sc[r + 8 * half][wave * 32 + 16 + col] = acc1[r];
  }
  __syncthreads();

  // softmax over T: row = tid>>4, 16 threads per row, 16 cols each
  int row = tid >> 4, sub = tid & 15;
  float mx = -3.4e38f;
#pragma unroll
  for (int i = 0; i < 16; ++i) mx = fmaxf(mx, sc[row][sub + 16 * i]);
  red[tid] = mx;
  __syncthreads();
  for (int s = 8; s > 0; s >>= 1) { if (sub < s) red[tid] = fmaxf(red[tid], red[tid + s]); __syncthreads(); }
  mx = red[row << 4];
  __syncthreads();

  float ssum = 0.0f;
#pragma unroll
  for (int i = 0; i < 16; ++i) {
    float e = __expf(sc[row][sub + 16 * i] - mx);
    sc[row][sub + 16 * i] = e;
    ssum += e;
  }
  red[tid] = ssum;
  __syncthreads();
  for (int s = 8; s > 0; s >>= 1) { if (sub < s) red[tid] += red[tid + s]; __syncthreads(); }
  float inv = 1.0f / red[row << 4];
  __syncthreads();
#pragma unroll
  for (int i = 0; i < 16; ++i) sc[row][sub + 16 * i] *= inv;
  __syncthreads();

  for (int e = tid; e < 16 * 256; e += 256) {
    int r = e >> 8, t = e & 255, gm = m0 + r;
    if (gm < NP) smbuf[((size_t)(b * NP + gm)) * NT + t] = sc[r][t];
  }
  {
    int t = tid;
    float s = 0.0f;
    for (int r = 0; r < 16; ++r) if (m0 + r < NP) s += sc[r][t];
    atomicAdd(&rel[b * NT + t], s);
  }
}

// ---------------------------------------------------------------------------
// Kernel 2: raters = rel >= mean(rel); denom = clip(#raters, 1e-5)
// ---------------------------------------------------------------------------
__global__ __launch_bounds__(256) void raters_kernel(const float* __restrict__ rel,
                                                     float* __restrict__ rater,
                                                     float* __restrict__ denom) {
  int b = blockIdx.x, t = threadIdx.x;
  __shared__ float red[256];
  float v = rel[b * NT + t] * (1.0f / NP);
  red[t] = v;
  __syncthreads();
  for (int s = 128; s > 0; s >>= 1) { if (t < s) red[t] += red[t + s]; __syncthreads(); }
  float mean = red[0] * (1.0f / NT);
  __syncthreads();
  float r = (v >= mean) ? 1.0f : 0.0f;
  rater[b * NT + t] = r;
  red[t] = r;
  __syncthreads();
  for (int s = 128; s > 0; s >>= 1) { if (t < s) red[t] += red[t + s]; __syncthreads(); }
  if (t == 0) denom[b] = fmaxf(red[0], 1e-5f);
}

// ---------------------------------------------------------------------------
// Kernel 3: sm_m (bf16) = sm * raters; vision_scores = sum_t(sm_m)/denom
// grid = (NP, B), block = 256
// ---------------------------------------------------------------------------
__global__ __launch_bounds__(256) void mask_kernel(const float* __restrict__ rater,
                                                   const float* __restrict__ denom,
                                                   const float* __restrict__ smf,
                                                   unsigned short* __restrict__ smb,
                                                   float* __restrict__ vscore) {
  int n = blockIdx.x, b = blockIdx.y, t = threadIdx.x;
  __shared__ float red[256];
  size_t off = ((size_t)(b * NP + n)) * NT + t;
  float v = smf[off] * rater[b * NT + t];
  smb[off] = f2bf(v);
  red[t] = v;
  __syncthreads();
  for (int s = 128; s > 0; s >>= 1) { if (t < s) red[t] += red[t + s]; __syncthreads(); }
  if (t == 0) vscore[b * NP + n] = red[0] / denom[b];
}

// ---------------------------------------------------------------------------
// Kernel 4: Gram matrix G = sm_m @ sm_mᵀ via WMMA, TDM-staged tiles.
// grid = (13*13, B), block = 32 (one wave per 16x16 tile)
// ---------------------------------------------------------------------------
__global__ __launch_bounds__(32) void gram_kernel(const unsigned short* __restrict__ smb,
                                                  float* __restrict__ G) {
  int b = blockIdx.y;
  int mi = blockIdx.x / 13, ni = blockIdx.x % 13;
  int lane = threadIdx.x;
  __shared__ alignas(16) unsigned short aS[2][16 * 32];
  __shared__ alignas(16) unsigned short bS[2][16 * 32];

  const unsigned short* pA = smb + (size_t)(b * NP + mi * 16) * NT;
  const unsigned short* pB = smb + (size_t)(b * NP + ni * 16) * NT;
  const int NSTG = NT / 32;  // 8
  v8f acc = {};

  if (lane == 0) {
    tdm_load_2d(lds_addr_of(aS[0]), pA, NT, NP - mi * 16, 32, 16, NT);
    tdm_load_2d(lds_addr_of(bS[0]), pB, NT, NP - ni * 16, 32, 16, NT);
  }
  for (int s = 0; s < NSTG; ++s) {
    int cur = s & 1;
    if (lane == 0) {
      if (s + 1 < NSTG) {
        int k0 = 32 * (s + 1);
        tdm_load_2d(lds_addr_of(aS[1 - cur]), pA + k0, NT - k0, NP - mi * 16, 32, 16, NT);
        tdm_load_2d(lds_addr_of(bS[1 - cur]), pB + k0, NT - k0, NP - ni * 16, 32, 16, NT);
        __builtin_amdgcn_s_wait_tensorcnt((short)2);
      } else {
        __builtin_amdgcn_s_wait_tensorcnt((short)0);
      }
    }
    __syncthreads();
    v16bf a  = load_frag_a(aS[cur], lane);
    v16bf bb = load_frag_b(bS[cur], 0, lane);
    acc = __builtin_amdgcn_wmma_f32_16x16x32_bf16(false, a, false, bb, (short)0, acc, false, false);
    __syncthreads();
  }
  int half = lane >> 4, col = lane & 15;
#pragma unroll
  for (int r = 0; r < 8; ++r) {
    int gm = mi * 16 + r + 8 * half, gn = ni * 16 + col;
    if (gm < NP && gn < NP) G[((size_t)b * NP + gm) * NP + gn] = acc[r];
  }
}

// ---------------------------------------------------------------------------
// Kernel 5: numerical rank via pivoted elimination on PSD Gram matrix.
// ---------------------------------------------------------------------------
__global__ __launch_bounds__(256) void rank_kernel(float* __restrict__ G,
                                                   int* __restrict__ keepK) {
  int b = blockIdx.x, tid = threadIdx.x;
  float* g = G + (size_t)b * NP * NP;
  __shared__ float red[256];
  __shared__ float piv, tol;
  __shared__ int rank;

  float mx = 0.0f;
  for (int i = tid; i < NP; i += 256) mx = fmaxf(mx, g[i * NP + i]);
  red[tid] = mx;
  __syncthreads();
  for (int s = 128; s > 0; s >>= 1) { if (tid < s) red[tid] = fmaxf(red[tid], red[tid + s]); __syncthreads(); }
  if (tid == 0) { tol = red[0] * 1e-10f + 1e-30f; rank = 0; }
  __syncthreads();

  for (int s = 0; s < NP; ++s) {
    if (tid == 0) { piv = g[s * NP + s]; if (piv > tol) rank++; }
    __syncthreads();
    float p = piv;
    if (p > tol) {
      float inv = 1.0f / p;
      int rem = NP - 1 - s;
      for (int e = tid; e < rem * rem; e += 256) {
        int i = s + 1 + e / rem;
        int j = s + 1 + e % rem;
        g[i * NP + j] -= g[i * NP + s] * g[s * NP + j] * inv;
      }
    }
    __syncthreads();
  }
  if (tid == 0) {
    int prune = (NP - rank) / 2;
    keepK[b] = NP - prune;
  }
}

// ---------------------------------------------------------------------------
// Kernel 6: per-batch top-K selection + recycle & cluster. grid=B, block=256.
// ---------------------------------------------------------------------------
__global__ __launch_bounds__(256) void select_kernel(const float* __restrict__ patch,
                                                     const float* __restrict__ vscore,
                                                     const int* __restrict__ keepK,
                                                     float* __restrict__ tokens,
                                                     float* __restrict__ recn_ws,
                                                     int* __restrict__ lenout,
                                                     int* __restrict__ tidx_ws) {
  int b = blockIdx.x, tid = threadIdx.x;
  __shared__ float sc[NP];
  __shared__ int kept[NP];
  __shared__ int dlist[NP];
  __shared__ float dsc[NP];
  __shared__ float rbuf[256];
  __shared__ int ibuf[256];
  __shared__ int rsel[64];
  __shared__ float mind[64];
  __shared__ int cent[32];
  __shared__ int asg[64];
  __shared__ int nprune;

  const float* vs = vscore + b * NP;
  for (int i = tid; i < NP; i += 256) { sc[i] = vs[i]; kept[i] = 0; }
  __syncthreads();
  int K = keepK[b];

  for (int k = 0; k < K; ++k) {
    float v = -3.4e38f; int idx = NP;
    if (tid < NP && !kept[tid]) { v = sc[tid]; idx = tid; }
    rbuf[tid] = v; ibuf[tid] = idx;
    __syncthreads();
    int pick = argmax_reduce(rbuf, ibuf, tid);
    if (tid == 0) { kept[pick] = 1; tidx_ws[b * NP + k] = pick; }
    __syncthreads();
  }

  for (int k = 0; k < K; ++k) {
    int idx = tidx_ws[b * NP + k];
    const float* src = patch + ((size_t)(b * NP + idx)) * ND;
    float* dst = tokens + ((size_t)(b * NP + k)) * ND;
    for (int d = tid; d < ND; d += 256) dst[d] = src[d];
  }

  if (tid == 0) {
    int c = 0;
    for (int n = 0; n < NP; ++n) if (!kept[n]) dlist[c++] = n;
    nprune = c;
  }
  __syncthreads();
  int prune = nprune;
  int nr = prune / 2;                    // TAU = 0.5, truncation
  if (nr < 1) { if (tid == 0) lenout[b] = K; return; }
  int nc = nr / 2; if (nc < 1) nc = 1;   // THETA = 0.5

  for (int i = tid; i < prune; i += 256) dsc[i] = sc[dlist[i]];
  __syncthreads();
  for (int k = 0; k < nr; ++k) {
    float v = -3.4e38f; int idx = NP;
    if (tid < prune) { v = dsc[tid]; idx = tid; }
    rbuf[tid] = v; ibuf[tid] = idx;
    __syncthreads();
    int j = argmax_reduce(rbuf, ibuf, tid);
    if (tid == 0) { dsc[j] = -3.4e38f; rsel[k] = dlist[j]; }
    __syncthreads();
  }

  float* rcn = recn_ws + (size_t)b * 64 * ND;
  for (int i = 0; i < nr; ++i) {
    const float* src = patch + ((size_t)(b * NP + rsel[i])) * ND;
    float ss = 0.0f;
    for (int d = tid; d < ND; d += 256) { float x = src[d]; ss += x * x; }
    rbuf[tid] = ss;
    __syncthreads();
    for (int s = 128; s > 0; s >>= 1) { if (tid < s) rbuf[tid] += rbuf[tid + s]; __syncthreads(); }
    float invn = 1.0f / fmaxf(sqrtf(rbuf[0]), 1e-12f);
    __syncthreads();
    for (int d = tid; d < ND; d += 256) rcn[(size_t)i * ND + d] = src[d] * invn;
  }
  __syncthreads();

  int wid = tid >> 5, lane = tid & 31;
  for (int j = wid; j < nr; j += 8) {
    float dsum = 0.0f;
    for (int d = lane; d < ND; d += 32) dsum += rcn[(size_t)j * ND + d] * rcn[d];
    for (int off = 16; off > 0; off >>= 1) dsum += __shfl_down(dsum, off);
    if (lane == 0) mind[j] = 1.0f - dsum;
  }
  if (tid == 0) cent[0] = 0;
  __syncthreads();

  for (int i = 1; i < nc; ++i) {
    float v = -3.4e38f; int idx = NP;
    if (tid < nr) { v = mind[tid]; idx = tid; }
    rbuf[tid] = v; ibuf[tid] = idx;
    __syncthreads();
    int cj = argmax_reduce(rbuf, ibuf, tid);
    if (tid == 0) cent[i] = cj;
    for (int j = wid; j < nr; j += 8) {
      float dsum = 0.0f;
      for (int d = lane; d < ND; d += 32) dsum += rcn[(size_t)j * ND + d] * rcn[(size_t)cj * ND + d];
      for (int off = 16; off > 0; off >>= 1) dsum += __shfl_down(dsum, off);
      if (lane == 0) mind[j] = fminf(mind[j], 1.0f - dsum);
    }
    __syncthreads();
  }

  for (int j = wid; j < nr; j += 8) {
    float best = -3.4e38f; int bi = 0;
    for (int c = 0; c < nc; ++c) {
      int cj = cent[c];
      float dsum = 0.0f;
      for (int d = lane; d < ND; d += 32) dsum += rcn[(size_t)j * ND + d] * rcn[(size_t)cj * ND + d];
      for (int off = 16; off > 0; off >>= 1) dsum += __shfl_down(dsum, off);
      dsum = __shfl(dsum, 0);
      if (dsum > best) { best = dsum; bi = c; }
    }
    if (lane == 0) asg[j] = bi;
  }
  __syncthreads();

  int outk = 0;
  for (int c = 0; c < nc; ++c) {
    int cnt = 0;
    for (int j = 0; j < nr; ++j) if (asg[j] == c) cnt++;
    if (cnt > 0) {
      float* dst = tokens + ((size_t)(b * NP + K + outk)) * ND;
      for (int d = tid; d < ND; d += 256) {
        float s = 0.0f;
        for (int j = 0; j < nr; ++j)
          if (asg[j] == c) s += patch[((size_t)(b * NP + rsel[j])) * ND + d];
        dst[d] = s;
      }
      outk++;
    }
  }
  if (tid == 0) lenout[b] = K + outk;
}

// ---------------------------------------------------------------------------
// Kernel 7: projection GEMM (bf16 WMMA, TDM-staged, double buffered).
// grid = (B, ceil(Lv/16), 3); block = 256.
// ---------------------------------------------------------------------------
__global__ __launch_bounds__(256) void proj_kernel(const unsigned short* __restrict__ tokb,
                                                   const unsigned short* __restrict__ wb,
                                                   const float* __restrict__ bias,
                                                   float* __restrict__ out,
                                                   int Lv, long bstride) {
  int b = blockIdx.x;
  int m0 = blockIdx.y * 16;
  int j0 = blockIdx.z * 256;
  int tid = threadIdx.x, wave = tid >> 5, lane = tid & 31;

  __shared__ alignas(16) unsigned short aS[2][16 * 32];
  __shared__ alignas(16) unsigned short bS[2][256 * 32];

  const unsigned short* pA = tokb + (size_t)(b * NP + m0) * ND;
  const unsigned short* pB = wb + (size_t)j0 * ND;
  const int NSTG = ND / 32;  // 24
  v8f acc0 = {}, acc1 = {};

  if (tid == 0) {
    tdm_load_2d(lds_addr_of(aS[0]), pA, ND, NP - m0, 32, 16, ND);
    tdm_load_2d(lds_addr_of(bS[0]), pB, ND, ND - j0, 32, 256, ND);
  }
  for (int s = 0; s < NSTG; ++s) {
    int cur = s & 1;
    if (tid == 0) {
      if (s + 1 < NSTG) {
        int k0 = 32 * (s + 1);
        tdm_load_2d(lds_addr_of(aS[1 - cur]), pA + k0, ND - k0, NP - m0, 32, 16, ND);
        tdm_load_2d(lds_addr_of(bS[1 - cur]), pB + k0, ND - k0, ND - j0, 32, 256, ND);
        __builtin_amdgcn_s_wait_tensorcnt((short)2);
      } else {
        __builtin_amdgcn_s_wait_tensorcnt((short)0);
      }
    }
    __syncthreads();
    v16bf a  = load_frag_a(aS[cur], lane);
    v16bf b0 = load_frag_b(bS[cur], wave * 32, lane);
    v16bf b1 = load_frag_b(bS[cur], wave * 32 + 16, lane);
    acc0 = __builtin_amdgcn_wmma_f32_16x16x32_bf16(false, a, false, b0, (short)0, acc0, false, false);
    acc1 = __builtin_amdgcn_wmma_f32_16x16x32_bf16(false, a, false, b1, (short)0, acc1, false, false);
    __syncthreads();
  }

  int half = lane >> 4, col = lane & 15;
#pragma unroll
  for (int r = 0; r < 8; ++r) {
    int gm = m0 + r + 8 * half;
    if (gm < Lv) {
      int j = j0 + wave * 32 + col;
      out[(size_t)b * bstride + (size_t)gm * ND + j]      = acc0[r] + bias[j];
      out[(size_t)b * bstride + (size_t)gm * ND + j + 16] = acc1[r] + bias[j + 16];
    }
  }
}

// ---------------------------------------------------------------------------
// Kernel 8: row LayerNorm (D = 768 = 3*256). grid = (rows, B); block = 256.
// ---------------------------------------------------------------------------
__global__ __launch_bounds__(256) void ln_rows_kernel(const float* __restrict__ in,
                                                      float* __restrict__ out,
                                                      const float* __restrict__ g,
                                                      const float* __restrict__ bb,
                                                      long in_bstride, long in_rstride,
                                                      long out_bstride, long out_rstride) {
  int m = blockIdx.x, b = blockIdx.y, t = threadIdx.x;
  const float* xi = in + (size_t)b * in_bstride + (size_t)m * in_rstride;
  float* xo = out + (size_t)b * out_bstride + (size_t)m * out_rstride;
  __shared__ float red[256];

  float vals[3];
  float s = 0.0f;
#pragma unroll
  for (int i = 0; i < 3; ++i) { vals[i] = xi[t + 256 * i]; s += vals[i]; }
  red[t] = s;
  __syncthreads();
  for (int q = 128; q > 0; q >>= 1) { if (t < q) red[t] += red[t + q]; __syncthreads(); }
  float mean = red[0] * (1.0f / ND);
  __syncthreads();
  float vv = 0.0f;
#pragma unroll
  for (int i = 0; i < 3; ++i) { float d = vals[i] - mean; vv += d * d; }
  red[t] = vv;
  __syncthreads();
  for (int q = 128; q > 0; q >>= 1) { if (t < q) red[t] += red[t + q]; __syncthreads(); }
  float inv = rsqrtf(red[0] * (1.0f / ND) + 1e-5f);
#pragma unroll
  for (int i = 0; i < 3; ++i) {
    int d = t + 256 * i;
    xo[d] = (vals[i] - mean) * inv * g[d] + bb[d];
  }
}

// ---------------------------------------------------------------------------
// Host launch
// ---------------------------------------------------------------------------
static inline size_t alignup(size_t x) { return (x + 255) & ~(size_t)255; }

extern "C" void kernel_launch(void* const* d_in, const int* in_sizes, int n_in,
                              void* d_out, int out_size, void* d_ws, size_t ws_size,
                              hipStream_t stream) {
  const float* patch  = (const float*)d_in[0];
  const float* text   = (const float*)d_in[1];
  const float* proj_w = (const float*)d_in[2];
  const float* proj_b = (const float*)d_in[3];
  const float* vis_g  = (const float*)d_in[4];
  const float* vis_b  = (const float*)d_in[5];
  const float* txt_g  = (const float*)d_in[6];
  const float* txt_b  = (const float*)d_in[7];
  float* out = (float*)d_out;

  int Lv = out_size / (NB * ND) - NT;   // padded visual length from output shape
  if (Lv < 0) Lv = 0;
  long S = (long)(Lv + NT) * ND;

  char* ws = (char*)d_ws;
  size_t off = 0;
  float* smbuf  = (float*)(ws + off); off = alignup(off + (size_t)NB * NP * NT * 4);
  unsigned short* smb16 = (unsigned short*)(ws + off); off = alignup(off + (size_t)NB * NP * NT * 2);
  float* rel    = (float*)(ws + off); off = alignup(off + (size_t)NB * NT * 4);
  float* rater  = (float*)(ws + off); off = alignup(off + (size_t)NB * NT * 4);
  float* denom  = (float*)(ws + off); off = alignup(off + (size_t)NB * 4);
  float* vscore = (float*)(ws + off); off = alignup(off + (size_t)NB * NP * 4);
  float* G      = (float*)(ws + off); off = alignup(off + (size_t)NB * NP * NP * 4);
  int*   keepK  = (int*)  (ws + off); off = alignup(off + (size_t)NB * 4);
  int*   lenbuf = (int*)  (ws + off); off = alignup(off + (size_t)NB * 4);
  int*   tidxbf = (int*)  (ws + off); off = alignup(off + (size_t)NB * NP * 4);
  float* tokens = (float*)(ws + off); off = alignup(off + (size_t)NB * NP * ND * 4);
  float* recn   = (float*)(ws + off); off = alignup(off + (size_t)NB * 64 * ND * 4);
  unsigned short* patchb = (unsigned short*)(ws + off); off = alignup(off + (size_t)NB * NP * ND * 2);
  unsigned short* textb  = (unsigned short*)(ws + off); off = alignup(off + (size_t)NB * NT * ND * 2);
  unsigned short* wbf    = (unsigned short*)(ws + off); off = alignup(off + (size_t)ND * ND * 2);
  unsigned short* tokb   = (unsigned short*)(ws + off); off = alignup(off + (size_t)NB * NP * ND * 2);

  // zero rel accumulator and token buffer (padding rows must stay 0)
  {
    size_t n1 = (size_t)NB * NT;
    zero_kernel<<<dim3((unsigned)((n1 + 255) / 256)), 256, 0, stream>>>(rel, n1);
    size_t n2 = (size_t)NB * NP * ND;
    zero_kernel<<<dim3((unsigned)((n2 + 255) / 256)), 256, 0, stream>>>(tokens, n2);
  }
  // pre-convert inputs to bf16 for TDM/WMMA staging
  {
    size_t np = (size_t)NB * NP * ND / 2;
    cvt_bf16_kernel<<<dim3((unsigned)((np + 255) / 256)), 256, 0, stream>>>(patch, (unsigned*)patchb, np);
    size_t nt = (size_t)NB * NT * ND / 2;
    cvt_bf16_kernel<<<dim3((unsigned)((nt + 255) / 256)), 256, 0, stream>>>(text, (unsigned*)textb, nt);
    size_t nw = (size_t)ND * ND / 2;
    cvt_bf16_kernel<<<dim3((unsigned)((nw + 255) / 256)), 256, 0, stream>>>(proj_w, (unsigned*)wbf, nw);
  }

  attn_kernel<<<dim3(NB, (NP + 15) / 16), 256, 0, stream>>>(patchb, textb, smbuf, rel);
  raters_kernel<<<NB, 256, 0, stream>>>(rel, rater, denom);
  mask_kernel<<<dim3(NP, NB), 256, 0, stream>>>(rater, denom, smbuf, smb16, vscore);
  gram_kernel<<<dim3(13 * 13, NB), 32, 0, stream>>>(smb16, G);
  rank_kernel<<<NB, 256, 0, stream>>>(G, keepK);
  select_kernel<<<NB, 256, 0, stream>>>(patch, vscore, keepK, tokens, recn, lenbuf, tidxbf);

  {
    size_t ntk = (size_t)NB * NP * ND / 2;
    cvt_bf16_kernel<<<dim3((unsigned)((ntk + 255) / 256)), 256, 0, stream>>>(tokens, (unsigned*)tokb, ntk);
  }

  if (Lv > 0) {
    int mt = (Lv + 15) / 16;
    proj_kernel<<<dim3(NB, mt, 3), 256, 0, stream>>>(tokb, wbf, proj_b, out, Lv, S);
    ln_rows_kernel<<<dim3(Lv, NB), 256, 0, stream>>>(out, out, vis_g, vis_b,
                                                     S, (long)ND, S, (long)ND);
  }
  ln_rows_kernel<<<dim3(NT, NB), 256, 0, stream>>>(text, out + (size_t)Lv * ND, txt_g, txt_b,
                                                   (long)NT * ND, (long)ND, S, (long)ND);
}
